// MultiheadAttentionWithSoftAdjacency_12524124635259
// MI455X (gfx1250) — matrix operand order repaired and checked by reading, not compiled, not run on hardware
//
#include <hip/hip_runtime.h>

// ---------------------------------------------------------------------------
// MultiheadAttentionWithSoftAdjacency for MI455X (gfx1250, wave32, WMMA)
// - all GEMMs on v_wmma_f32_16x16x32_f16 (f16 in, f32 accumulate)
// - attention GEMMs: 64x64 block tiles, B double-buffered through LDS with
//   global_load_async_to_lds_b128 (ASYNCcnt) overlapping the WMMA bursts
// - attn [B,H,S,S] f32 materialized directly in its d_out slice
// ---------------------------------------------------------------------------

typedef _Float16 v16h __attribute__((ext_vector_type(16)));
typedef _Float16 v8h  __attribute__((ext_vector_type(8)));
typedef float    v8f  __attribute__((ext_vector_type(8)));

#define DEV __device__ __forceinline__

constexpr int B_  = 2;
constexpr int S_  = 2048;
constexpr int D_  = 128;
constexpr int H_  = 8;
constexpr int HD_ = 1024;          // H*D
constexpr int BS_ = B_ * S_;       // 4096 rows
constexpr float ALPHA_ = 0.5f;
constexpr float EPS_   = 1e-5f;
constexpr float SCALE_ = 0.08838834764831845f; // 1/sqrt(128)

DEV v8f wmma_f16(v16h a, v16h b, v8f c) {
  return __builtin_amdgcn_wmma_f32_16x16x32_f16(false, a, false, b, (short)0, c,
                                                false, false);
}

// ---- CDNA5 async copy helpers ---------------------------------------------

// Low 32 bits of a generic LDS pointer are the wave-relative LDS byte address
// (flat->LDS aperture truncates to addr[31:0], per ISA 10.2).
DEV unsigned lds_addr32(const void* p) { return (unsigned)(uintptr_t)p; }

// GLOBAL_LOAD_ASYNC_TO_LDS_B128: VDST = per-lane LDS address VGPR,
// VADDR = 64-bit global address, no SADDR. Tracked with ASYNCcnt.
DEV void async_copy_b128(unsigned lds, const void* gaddr) {
  asm volatile("global_load_async_to_lds_b128 %0, %1, off"
               :: "v"(lds), "v"(gaddr)
               : "memory");
}

DEV void wait_async0() {
#if __has_builtin(__builtin_amdgcn_s_wait_asynccnt)
  __builtin_amdgcn_s_wait_asynccnt(0);
#else
  asm volatile("s_wait_asynccnt 0x0" ::: "memory");
#endif
}

// ---- WMMA operand loaders (contiguous k = 16h + s order, wave32) ----------

// A 16x32 f16 tile from row-major [*, lda]: lane L<16 -> M=L, K=0..15;
// lane L>=16 -> M=L-16, K=16..31. Two contiguous 8-half (16B) loads.
DEV v16h load_a_f16(const _Float16* __restrict__ A, int lda, int m0, int k0, int lane) {
  int m  = m0 + (lane & 15);
  int kb = k0 + ((lane >> 4) << 4);
  const _Float16* p = A + (size_t)m * lda + kb;
  v8h lo = *(const v8h*)(p);
  v8h hi = *(const v8h*)(p + 8);
  v16h r;
#pragma unroll
  for (int i = 0; i < 8; ++i) { r[i] = lo[i]; r[8 + i] = hi[i]; }
  return r;
}

// Same A tile but converting from f32 source (attn rows).
DEV v16h load_a_f32(const float* __restrict__ A, int lda, int m0, int k0, int lane) {
  int m  = m0 + (lane & 15);
  int kb = k0 + ((lane >> 4) << 4);
  const float* p = A + (size_t)m * lda + kb;
  v16h r;
#pragma unroll
  for (int i = 0; i < 8; ++i) {
    r[i]     = (_Float16)p[i];
    r[8 + i] = (_Float16)p[8 + i];
  }
  return r;
}

// B 32x16 tile pre-packed as [tile][lane][16 halfs] -> one 32B load per lane.
DEV v16h load_b_packed(const _Float16* __restrict__ Bp, int tileIdx, int lane) {
  return *(const v16h*)(Bp + ((size_t)tileIdx * 32 + lane) * 16);
}

// ---- elementwise / packing kernels ----------------------------------------

__global__ void cvt_f16_kernel(const float* __restrict__ s, _Float16* __restrict__ d, int n) {
  int i = blockIdx.x * blockDim.x + threadIdx.x;
  if (i < n) d[i] = (_Float16)s[i];
}

// Pack row-major f32 W[K,N] into WMMA B-operand layout (f16).
__global__ void pack_b_kernel(const float* __restrict__ W, _Float16* __restrict__ P,
                              int K, int N) {
  int idx = blockIdx.x * blockDim.x + threadIdx.x;
  if (idx >= K * N) return;
  int k = idx / N, n = idx % N;
  int kt = k >> 5, kin = k & 31, nt = n >> 4, nin = n & 15;
  int lanep = ((kin >> 4) << 4) | nin;
  size_t flat = ((size_t)(kt * (N >> 4) + nt) * 32 + lanep) * 16 + (kin & 15);
  P[flat] = (_Float16)W[idx];
}

// ---- generic GEMM: C[M,N] = A[M,K] @ Bpacked + bias -----------------------
// Each wave computes a 16 x (16*NT) strip: A loaded once per k-chunk, all NT
// B tiles loaded before the WMMA burst. OUTMODE: 0=f32 RM, 1=f16 RM, 2=packed.
template <int OUTMODE, bool RELU, int NT>
__global__ void gemm_f16_kernel(const _Float16* __restrict__ A, int lda,
                                const _Float16* __restrict__ Bp,
                                const float* __restrict__ bias,
                                int Mtiles, int Ntiles, int Ksize,
                                float* __restrict__ outF32,
                                _Float16* __restrict__ outF16,
                                _Float16* __restrict__ outPk,
                                int ldo) {
  int ngrp = Ntiles / NT;
  int wave = blockIdx.x * (blockDim.x >> 5) + (threadIdx.x >> 5);
  if (wave >= Mtiles * ngrp) return;            // wave-uniform
  int lane = threadIdx.x & 31;
  int mt = wave / ngrp, ng = wave % ngrp;
  int m0 = mt * 16, nt0 = ng * NT;

  v8f acc[NT] = {};
  for (int k0 = 0; k0 < Ksize; k0 += 32) {
    v16h a = load_a_f16(A, lda, m0, k0, lane);
    v16h b[NT];
#pragma unroll
    for (int j = 0; j < NT; ++j)
      b[j] = load_b_packed(Bp, (k0 >> 5) * Ntiles + nt0 + j, lane);
#pragma unroll
    for (int j = 0; j < NT; ++j)
      acc[j] = wmma_f16(a, b[j], acc[j]);
  }

  int hi = lane >> 4;
#pragma unroll
  for (int j = 0; j < NT; ++j) {
    int n = (nt0 + j) * 16 + (lane & 15);
    float bn = bias ? bias[n] : 0.0f;
#pragma unroll
    for (int r = 0; r < 8; ++r) {
      int m = m0 + r + hi * 8;
      float v = acc[j][r] + bn;
      if (RELU) v = fmaxf(v, 0.0f);
      if (OUTMODE == 0) {
        outF32[(size_t)m * ldo + n] = v;
      } else if (OUTMODE == 1) {
        outF16[(size_t)m * ldo + n] = (_Float16)v;
      } else {
        int kt = m >> 5, kin = m & 31, ntp = n >> 4, nin = n & 15;
        int lanep = ((kin >> 4) << 4) | nin;
        size_t flat = ((size_t)(kt * (ldo >> 4) + ntp) * 32 + lanep) * 16 + (kin & 15);
        outPk[flat] = (_Float16)v;
      }
    }
  }
}

// ---- scores: 64x64 block tile, K^T double-buffered via async copy ---------
// block = 128 thr (4 waves). wave w owns rows [qblk*64 + w*16, +16) and all
// four 16-col tiles of the strip; each wave stages one B tile per chunk.
// Pipeline per chunk t: wait(t) -> barrier -> issue async(t+1) -> compute(t).
// The top barrier also proves all waves finished reading the buffer that
// async(t+1) overwrites, so one barrier per iteration suffices.
__global__ void scores_kernel(const _Float16* __restrict__ qh,
                              const _Float16* __restrict__ kh,
                              const int* __restrict__ adj,
                              float* __restrict__ attn) {
  __shared__ __align__(32) _Float16 ldsB[2][4 * 32 * 16];   // 2 x 4 KB
  int bid = blockIdx.x;
  int kblk = bid % (S_ / 64); bid /= (S_ / 64);
  int qblk = bid % (S_ / 64); bid /= (S_ / 64);
  int h = bid % H_; int b = bid / H_;
  int tid = threadIdx.x, lane = tid & 31, wid = tid >> 5;

  const _Float16* qb = qh + (size_t)b * S_ * HD_ + h * D_;
  const _Float16* kb = kh + (size_t)b * S_ * HD_ + h * D_;
  int m0 = qblk * 64 + wid * 16;

  auto stage = [&](int buf, int kk) {
    int n  = kblk * 64 + wid * 16 + (lane & 15);
    int ko = kk + ((lane >> 4) << 4);
    const _Float16* src = kb + (size_t)n * HD_ + ko;
    unsigned dst = lds_addr32(&ldsB[buf][(wid * 32 + lane) * 16]);
    async_copy_b128(dst, src);
    async_copy_b128(dst + 16, src + 8);
  };

  stage(0, 0);
  v8f acc[4] = {};
  constexpr int NCH = D_ / 32;                   // 4 k-chunks
  for (int t = 0; t < NCH; ++t) {
    int cur = t & 1;
    wait_async0();
    __syncthreads();
    if (t + 1 < NCH) stage(cur ^ 1, (t + 1) * 32);
    v16h a = load_a_f16(qb, HD_, m0, t * 32, lane);
    v16h bm[4];
#pragma unroll
    for (int j = 0; j < 4; ++j)
      bm[j] = *(const v16h*)(&ldsB[cur][(j * 32 + lane) * 16]);
#pragma unroll
    for (int j = 0; j < 4; ++j)
      acc[j] = wmma_f16(a, bm[j], acc[j]);
  }

  float* arow = attn + (size_t)(b * H_ + h) * S_ * S_;
  int hi = lane >> 4;
#pragma unroll
  for (int j = 0; j < 4; ++j) {
    int n = kblk * 64 + j * 16 + (lane & 15);
#pragma unroll
    for (int r = 0; r < 8; ++r) {
      int qi = m0 + r + hi * 8;
      float mask = ALPHA_ * (float)adj[(size_t)qi * S_ + n] + (1.0f - ALPHA_);
      arow[(size_t)qi * S_ + n] = acc[j][r] * SCALE_ * mask;
    }
  }
}

// ---- in-place row softmax over S=2048, one block (256 thr) per row --------
__global__ void softmax_kernel(float* __restrict__ attn) {
  __shared__ float red[256];
  float* p = attn + (size_t)blockIdx.x * S_;
  int t = threadIdx.x;
  float v[8];
  float mx = -3.4e38f;
#pragma unroll
  for (int j = 0; j < 8; ++j) { v[j] = p[t + j * 256]; mx = fmaxf(mx, v[j]); }
  red[t] = mx; __syncthreads();
  for (int s = 128; s > 0; s >>= 1) { if (t < s) red[t] = fmaxf(red[t], red[t + s]); __syncthreads(); }
  mx = red[0]; __syncthreads();
  float sum = 0.0f;
#pragma unroll
  for (int j = 0; j < 8; ++j) { v[j] = __expf(v[j] - mx); sum += v[j]; }
  red[t] = sum; __syncthreads();
  for (int s = 128; s > 0; s >>= 1) { if (t < s) red[t] += red[t + s]; __syncthreads(); }
  float inv = 1.0f / red[0];
#pragma unroll
  for (int j = 0; j < 8; ++j) p[t + j * 256] = v[j] * inv;
}

// ---- attn @ V: 64x64 block tile, packed V double-buffered via async -------
__global__ void attnv_kernel(const float* __restrict__ attn,
                             const _Float16* __restrict__ vp,
                             _Float16* __restrict__ attout) {
  __shared__ __align__(32) _Float16 ldsB[2][4 * 32 * 16];
  int bid = blockIdx.x;
  int nblk = bid % 2;          bid /= 2;          // which 64-col half of head
  int qblk = bid % (S_ / 64);  bid /= (S_ / 64);
  int h = bid % H_; int b = bid / H_;
  int tid = threadIdx.x, lane = tid & 31, wid = tid >> 5;

  const float* arow = attn + (size_t)(b * H_ + h) * S_ * S_;
  int m0 = qblk * 64 + wid * 16;

  auto stage = [&](int buf, int k0) {
    int tileIdx = (b * (S_ / 32) + (k0 >> 5)) * (HD_ / 16)
                + h * (D_ / 16) + nblk * 4 + wid;
    const _Float16* src = vp + ((size_t)tileIdx * 32 + lane) * 16;
    unsigned dst = lds_addr32(&ldsB[buf][(wid * 32 + lane) * 16]);
    async_copy_b128(dst, src);
    async_copy_b128(dst + 16, src + 8);
  };

  stage(0, 0);
  v8f acc[4] = {};
  constexpr int NCH = S_ / 32;                   // 64 k-chunks
  for (int t = 0; t < NCH; ++t) {
    int cur = t & 1;
    wait_async0();
    __syncthreads();
    if (t + 1 < NCH) {
      stage(cur ^ 1, (t + 1) * 32);
      int m = m0 + (lane & 15);
      __builtin_prefetch(arow + (size_t)m * S_ + (t + 1) * 32, 0, 0);
    }
    v16h a = load_a_f32(arow, S_, m0, t * 32, lane);   // cvt f32->f16, x4 reuse
    v16h bm[4];
#pragma unroll
    for (int j = 0; j < 4; ++j)
      bm[j] = *(const v16h*)(&ldsB[cur][(j * 32 + lane) * 16]);
#pragma unroll
    for (int j = 0; j < 4; ++j)
      acc[j] = wmma_f16(a, bm[j], acc[j]);
  }

  int hi = lane >> 4;
#pragma unroll
  for (int j = 0; j < 4; ++j) {
    int n = h * D_ + nblk * 64 + j * 16 + (lane & 15);
#pragma unroll
    for (int r = 0; r < 8; ++r) {
      int q = m0 + r + hi * 8;
      attout[(size_t)(b * S_ + q) * HD_ + n] = (_Float16)acc[j][r];
    }
  }
}

// ---- y = LayerNorm(a + c) * g + be ; optional f32 / f16 outputs -----------
__global__ void add_ln_kernel(const float* __restrict__ a, const float* __restrict__ c,
                              const float* __restrict__ g, const float* __restrict__ be,
                              float* __restrict__ outF32, _Float16* __restrict__ outF16) {
  __shared__ float red[128];
  int row = blockIdx.x, t = threadIdx.x;
  float v = a[(size_t)row * D_ + t] + c[(size_t)row * D_ + t];
  red[t] = v; __syncthreads();
  for (int s = 64; s > 0; s >>= 1) { if (t < s) red[t] += red[t + s]; __syncthreads(); }
  float mu = red[0] * (1.0f / D_); __syncthreads();
  float d = v - mu;
  red[t] = d * d; __syncthreads();
  for (int s = 64; s > 0; s >>= 1) { if (t < s) red[t] += red[t + s]; __syncthreads(); }
  float rstd = rsqrtf(red[0] * (1.0f / D_) + EPS_);
  float y = d * rstd * g[t] + be[t];
  if (outF32) outF32[(size_t)row * D_ + t] = y;
  if (outF16) outF16[(size_t)row * D_ + t] = (_Float16)y;
}

// ---------------------------------------------------------------------------

extern "C" void kernel_launch(void* const* d_in, const int* in_sizes, int n_in,
                              void* d_out, int out_size, void* d_ws, size_t ws_size,
                              hipStream_t stream) {
  const float* x   = (const float*)d_in[0];
  const int*   adj = (const int*)d_in[1];
  const float* Wq  = (const float*)d_in[2];
  const float* bq  = (const float*)d_in[3];
  const float* Wk  = (const float*)d_in[4];
  const float* bk  = (const float*)d_in[5];
  const float* Wv  = (const float*)d_in[6];
  const float* bv  = (const float*)d_in[7];
  const float* Wo  = (const float*)d_in[8];
  const float* bo  = (const float*)d_in[9];
  const float* g1  = (const float*)d_in[10];
  const float* be1 = (const float*)d_in[11];
  const float* g2  = (const float*)d_in[12];
  const float* be2 = (const float*)d_in[13];
  const float* W1  = (const float*)d_in[14];
  const float* b1  = (const float*)d_in[15];
  const float* W2  = (const float*)d_in[16];
  const float* b2  = (const float*)d_in[17];

  float* y    = (float*)d_out;                       // [B,S,D]
  float* attn = y + (size_t)BS_ * D_;                // [B,H,S,S] output slice

  size_t off = 0;
  auto carve = [&](size_t bytes) -> void* {
    void* p = (char*)d_ws + off;
    off = (off + bytes + 255) & ~(size_t)255;
    return p;
  };
  _Float16* xh    = (_Float16*)carve((size_t)BS_ * D_ * 2);
  _Float16* qh    = (_Float16*)carve((size_t)BS_ * HD_ * 2);
  _Float16* kh    = (_Float16*)carve((size_t)BS_ * HD_ * 2);
  _Float16* vp    = (_Float16*)carve((size_t)BS_ * HD_ * 2);   // B-packed V
  _Float16* atth  = (_Float16*)carve((size_t)BS_ * HD_ * 2);
  _Float16* hf16  = (_Float16*)carve((size_t)BS_ * D_ * 2);
  _Float16* ff1h  = (_Float16*)carve((size_t)BS_ * D_ * 2);
  float*    oproj = (float*)carve((size_t)BS_ * D_ * 4);
  float*    hf32  = (float*)carve((size_t)BS_ * D_ * 4);
  float*    ff2f  = (float*)carve((size_t)BS_ * D_ * 4);
  _Float16* Wqp   = (_Float16*)carve((size_t)D_ * HD_ * 2);
  _Float16* Wkp   = (_Float16*)carve((size_t)D_ * HD_ * 2);
  _Float16* Wvp   = (_Float16*)carve((size_t)D_ * HD_ * 2);
  _Float16* Wop   = (_Float16*)carve((size_t)HD_ * D_ * 2);
  _Float16* W1p   = (_Float16*)carve((size_t)D_ * D_ * 2);
  _Float16* W2p   = (_Float16*)carve((size_t)D_ * D_ * 2);
  (void)ws_size; (void)in_sizes; (void)n_in; (void)out_size;

  // 1) precision conversion / weight packing
  cvt_f16_kernel<<<(BS_ * D_ + 255) / 256, 256, 0, stream>>>(x, xh, BS_ * D_);
  pack_b_kernel<<<(D_ * HD_ + 255) / 256, 256, 0, stream>>>(Wq, Wqp, D_, HD_);
  pack_b_kernel<<<(D_ * HD_ + 255) / 256, 256, 0, stream>>>(Wk, Wkp, D_, HD_);
  pack_b_kernel<<<(D_ * HD_ + 255) / 256, 256, 0, stream>>>(Wv, Wvp, D_, HD_);
  pack_b_kernel<<<(HD_ * D_ + 255) / 256, 256, 0, stream>>>(Wo, Wop, HD_, D_);
  pack_b_kernel<<<(D_ * D_ + 255) / 256, 256, 0, stream>>>(W1, W1p, D_, D_);
  pack_b_kernel<<<(D_ * D_ + 255) / 256, 256, 0, stream>>>(W2, W2p, D_, D_);

  // 2) QKV projections: [4096,128] @ [128,1024], 16x64 strip per wave
  {
    int Mtiles = BS_ / 16, Ntiles = HD_ / 16;          // 256 x 64
    int waves  = Mtiles * (Ntiles / 4);
    int blocks = (waves + 3) / 4;
    gemm_f16_kernel<1, false, 4><<<blocks, 128, 0, stream>>>(
        xh, D_, Wqp, bq, Mtiles, Ntiles, D_, nullptr, qh, nullptr, HD_);
    gemm_f16_kernel<1, false, 4><<<blocks, 128, 0, stream>>>(
        xh, D_, Wkp, bk, Mtiles, Ntiles, D_, nullptr, kh, nullptr, HD_);
    gemm_f16_kernel<2, false, 4><<<blocks, 128, 0, stream>>>(
        xh, D_, Wvp, bv, Mtiles, Ntiles, D_, nullptr, nullptr, vp, HD_);
  }

  // 3) masked scores -> attn (f32, direct to output slice)
  scores_kernel<<<B_ * H_ * (S_ / 64) * (S_ / 64), 128, 0, stream>>>(qh, kh, adj, attn);

  // 4) row softmax in place
  softmax_kernel<<<B_ * H_ * S_, 256, 0, stream>>>(attn);

  // 5) attn @ V -> att_out (f16, [4096,1024])
  attnv_kernel<<<B_ * H_ * (S_ / 64) * 2, 128, 0, stream>>>(attn, vp, atth);

  // 6) out projection: [4096,1024] @ [1024,128] + bo -> f32
  {
    int Mtiles = BS_ / 16, Ntiles = D_ / 16;           // 256 x 8
    int waves  = Mtiles * (Ntiles / 4);
    int blocks = (waves + 3) / 4;
    gemm_f16_kernel<0, false, 4><<<blocks, 128, 0, stream>>>(
        atth, HD_, Wop, bo, Mtiles, Ntiles, HD_, oproj, nullptr, nullptr, D_);
  }

  // 7) h = LN(x + out) -> f32 + f16
  add_ln_kernel<<<BS_, D_, 0, stream>>>(x, oproj, g1, be1, hf32, hf16);

  // 8) FFN: relu(h@W1+b1) @ W2 + b2
  {
    int Mtiles = BS_ / 16, Ntiles = D_ / 16;
    int waves  = Mtiles * (Ntiles / 4);
    int blocks = (waves + 3) / 4;
    gemm_f16_kernel<1, true, 4><<<blocks, 128, 0, stream>>>(
        hf16, D_, W1p, b1, Mtiles, Ntiles, D_, nullptr, ff1h, nullptr, D_);
    gemm_f16_kernel<0, false, 4><<<blocks, 128, 0, stream>>>(
        ff1h, D_, W2p, b2, Mtiles, Ntiles, D_, ff2f, nullptr, nullptr, D_);
  }

  // 9) y = LN(h + ff) -> d_out
  add_ln_kernel<<<BS_, D_, 0, stream>>>(hf32, ff2f, g2, be2, y, nullptr);
}
